// ResnetEncoderMatching_66829691125931
// MI455X (gfx1250) — compile-verified, matrix operand order checked by reading, not run on hardware
//
#include <hip/hip_runtime.h>
#include <hip/hip_bf16.h>

typedef __attribute__((ext_vector_type(16))) _Float16 v16h;
typedef __attribute__((ext_vector_type(8)))  _Float16 v8h;
typedef __attribute__((ext_vector_type(8)))  float    v8f;
typedef unsigned int v4u __attribute__((ext_vector_type(4)));
typedef int          v8i __attribute__((ext_vector_type(8)));
typedef int          v4i __attribute__((ext_vector_type(4)));

#define IMG_H 192
#define IMG_W 640
#define F0_H 96
#define F0_W 320
#define MH 48
#define MW 160
#define CCH 64
#define NIMG 4
#define NBINS 80

// padded activation layout: [n][PH][PADW][64] f16, 1-pixel zero halo
#define PH   50
#define PADW 162
#define PIMG ((size_t)PH * PADW * CCH)          // halfs per image
#define PBUF_BYTES (NIMG * PIMG * 2)            // 4,147,200 B per buffer

// packed weight image: [K2=18][kb=2][nc=64][e=16] f16; chunk = 18432 halfs
#define PW_ELEMS 36864
#define CHUNK_HALFS 18432                       /* 36864 bytes */

// ---------------------------------------------------------------------------
// Kernel Z: zero the three padded ping-pong buffers (vector stores)
// ---------------------------------------------------------------------------
__global__ __launch_bounds__(256)
void zero_bufs(v8h* __restrict__ p, long long nvec)
{
    const long long i = (long long)blockIdx.x * 256 + threadIdx.x;
    if (i >= nvec) return;
    const v8h z = {};
    p[i] = z;
}

// ---------------------------------------------------------------------------
// Kernel 0: pre-swizzle conv3x3 weights (f32 OIHW) into the per-lane WMMA
// B-fragment image (f16): element (K2,kb,nc,e) <- w[nc][ci][ky][kx],
// k = K2*32 + kb*8 + (e<8?e:e+8)
// ---------------------------------------------------------------------------
__global__ __launch_bounds__(256)
void pack_weights(const float* __restrict__ wgt, _Float16* __restrict__ pw)
{
    const int i = blockIdx.x * 256 + threadIdx.x;
    if (i >= PW_ELEMS) return;
    const int e  = i & 15;
    const int nc = (i >> 4) & 63;
    const int kb = (i >> 10) & 1;
    const int K2 = i >> 11;                      // 0..17
    const int k  = K2 * 32 + kb * 8 + (e < 8 ? e : e + 8);
    const int tap = k >> 6;                      // ky*3+kx
    const int ci  = k & 63;
    pw[i] = (_Float16)wgt[((nc * 64 + ci) * 3 + tap / 3) * 3 + tap % 3];
}

// ---------------------------------------------------------------------------
// Kernel 1: conv7x7 stride2 pad3 (3->64) + BN + ReLU, fused input norm
// out: f16 NHWC [4,96,320,64] (unpadded)
// ---------------------------------------------------------------------------
__global__ __launch_bounds__(256)
void conv1_bn_relu(const float* __restrict__ cur, const float* __restrict__ lkp,
                   const float* __restrict__ w,
                   const float* __restrict__ bg, const float* __restrict__ bb,
                   const float* __restrict__ bm, const float* __restrict__ bv,
                   _Float16* __restrict__ f0h)
{
    __shared__ float Wl[64 * 3 * 49];   // 9408 floats
    const int tid = threadIdx.x;
    for (int i = tid; i < 9408; i += 256) Wl[i] = w[i];
    __syncthreads();

    const int o  = tid & 63;
    const int px = tid >> 6;                 // 0..3
    const int x  = blockIdx.x * 4 + px;      // 0..319
    const int y  = blockIdx.y;               // 0..95
    const int n  = blockIdx.z;               // 0..3

    const float* img = (n < 2) ? (cur + (size_t)n * 3 * IMG_H * IMG_W)
                               : (lkp + (size_t)(n - 2) * 3 * IMG_H * IMG_W);

    float sum = 0.0f;
    for (int ci = 0; ci < 3; ++ci) {
        for (int ky = 0; ky < 7; ++ky) {
            const int iy = y * 2 - 3 + ky;
            if (iy < 0 || iy >= IMG_H) continue;
            for (int kx = 0; kx < 7; ++kx) {
                const int ix = x * 2 - 3 + kx;
                if (ix < 0 || ix >= IMG_W) continue;
                const float val = (img[((size_t)ci * IMG_H + iy) * IMG_W + ix] - 0.45f) * (1.0f / 0.225f);
                sum += val * Wl[((o * 3 + ci) * 7 + ky) * 7 + kx];
            }
        }
    }
    const float sc = bg[o] * rsqrtf(bv[o] + 1e-5f);
    float val = sum * sc + (bb[o] - bm[o] * sc);
    val = fmaxf(val, 0.0f);
    f0h[(((size_t)n * F0_H + y) * F0_W + x) * CCH + o] = (_Float16)val;
}

// ---------------------------------------------------------------------------
// Kernel 2: maxpool 3x3 stride2 pad1 -> PADDED f16 [4,50,162,64] interior
// ---------------------------------------------------------------------------
__global__ __launch_bounds__(256)
void maxpool3x3s2(const _Float16* __restrict__ f0h, _Float16* __restrict__ x0h)
{
    const long long idx = (long long)blockIdx.x * 256 + threadIdx.x;
    const long long total = (long long)NIMG * MH * MW * CCH;
    if (idx >= total) return;
    const int c = (int)(idx & 63);
    long long t = idx >> 6;
    const int x = (int)(t % MW); t /= MW;
    const int y = (int)(t % MH);
    const int n = (int)(t / MH);
    float best = -1e30f;
    for (int dy = -1; dy <= 1; ++dy) {
        const int iy = 2 * y + dy;
        if (iy < 0 || iy >= F0_H) continue;
        for (int dx = -1; dx <= 1; ++dx) {
            const int ix = 2 * x + dx;
            if (ix < 0 || ix >= F0_W) continue;
            const float v = (float)f0h[(((size_t)n * F0_H + iy) * F0_W + ix) * CCH + c];
            best = fmaxf(best, v);
        }
    }
    x0h[((size_t)n * PH + (y + 1)) * PADW * CCH + (size_t)(x + 1) * CCH + c] = (_Float16)best;
}

// ---------------------------------------------------------------------------
// Kernel 3: implicit-GEMM conv3x3 (64->64) on PADDED activations via
// v_wmma_f32_16x16x32_f16. Grid (10,48,4): y uniform per block, M-tile = 16
// pixels along x (never crosses a row), so all tap offsets are immediates.
// Weights pre-swizzled, staged to LDS with the Tensor Data Mover.
// ---------------------------------------------------------------------------
__global__ __launch_bounds__(128)
void conv3x3_wmma_bn(const _Float16* __restrict__ in,   // padded NHWC f16
                     const _Float16* __restrict__ pw,   // packed weights (36864 f16)
                     const float* __restrict__ bg, const float* __restrict__ bb,
                     const float* __restrict__ bm, const float* __restrict__ bv,
                     const _Float16* __restrict__ res,  // nullable residual (padded)
                     _Float16* __restrict__ out)        // padded NHWC f16
{
    __shared__ alignas(16) _Float16 Bl[CHUNK_HALFS];    // 36 KB, one K-chunk

    const int tid  = threadIdx.x;
    const int lane = tid & 31;
    const int wave = tid >> 5;           // N-tile index (0..3)
    const int xt   = blockIdx.x;         // 0..9
    const int y    = blockIdx.y;         // 0..47 (uniform per block)
    const int n    = blockIdx.z;         // image 0..3

    const int mrow = lane & 15;
    const int x    = xt * 16 + mrow;             // A-matrix row = pixel along x
    const int kb   = lane >> 4;                  // kbase = kb*8 per ISA A layout
    const int nc   = wave * 16 + mrow;           // output channel (B col / epilogue)

    // always-in-bounds base (zero halo handles conv padding)
    const _Float16* abase =
        in + ((size_t)n * PH + (y + 1)) * PADW * CCH + (size_t)(x + 1) * CCH + kb * 8;
    const _Float16* bbase = &Bl[(kb * 64 + nc) * 16];

    v8f acc = {};
    for (int chunk = 0; chunk < 2; ++chunk) {
        const _Float16* src = pw + (size_t)chunk * CHUNK_HALFS;
#if defined(__gfx1250__) && __has_builtin(__builtin_amdgcn_tensor_load_to_lds)
        if (wave == 0) {
            // D# group0/group1 (CDNA5 ISA): one valid descriptor,
            // type=2, 4-byte elements, 2-D tile 9216x1 dwords (36864 B).
            const unsigned long long ga = (unsigned long long)(const void*)src;
            const unsigned int lds = (unsigned int)(unsigned long long)(const void*)&Bl[0];
            v4u g0;
            g0[0] = 1u;                                          // count=1 (valid)
            g0[1] = lds;                                         // lds_addr
            g0[2] = (unsigned int)(ga & 0xffffffffu);            // global_addr[31:0]
            g0[3] = (unsigned int)((ga >> 32) & 0x1ffffffu) | (2u << 30); // [56:32] + type=2
            v8i g1;
            g1[0] = (2 << 16);                                   // data_size=4B
            g1[1] = (int)(9216u << 16);                          // tensor_dim0 lo16
            g1[2] = (int)(1u << 16);                             // dim0 hi=0, tensor_dim1=1
            g1[3] = (int)(9216u << 16);                          // dim1 hi=0, tile_dim0=9216
            g1[4] = 1;                                           // tile_dim1=1, tile_dim2=0
            g1[5] = 9216;                                        // tensor_dim0_stride
            g1[6] = 0;
            g1[7] = 0;
            v4i gz = {};
#if __clang_major__ >= 23
            v8i gz8 = {};
            __builtin_amdgcn_tensor_load_to_lds(g0, g1, gz, gz, gz8, 0);
#else
            __builtin_amdgcn_tensor_load_to_lds(g0, g1, gz, gz, 0);
#endif
            __builtin_amdgcn_s_wait_tensorcnt(0);
        }
#else
        for (int i = tid; i < CHUNK_HALFS / 8; i += 128)
            ((v8h*)Bl)[i] = ((const v8h*)src)[i];
#endif
        __syncthreads();

#pragma unroll
        for (int ks = 0; ks < 9; ++ks) {
            const int kstep = chunk * 9 + ks;
            const int tap   = kstep >> 1;            // 32 K-vals = half a tap's 64 ch
            const int dy    = tap / 3 - 1;
            const int dx    = tap % 3 - 1;
            const int cbase = (kstep & 1) * 32;

            // A fragment: unconditional 16B loads at compile-time immediate offsets
            const _Float16* ap = abase + (dy * PADW + dx) * CCH + cbase;
            const v8h alo = *(const v8h*)(ap);
            const v8h ahi = *(const v8h*)(ap + 16);
            const v16h a = __builtin_shufflevector(alo, ahi,
                             0, 1, 2, 3, 4, 5, 6, 7, 8, 9, 10, 11, 12, 13, 14, 15);

            // B fragment: pre-swizzled, 32B contiguous per lane in LDS
            const v8h* bp = (const v8h*)(bbase + ks * 2048);
            const v16h b = __builtin_shufflevector(bp[0], bp[1],
                             0, 1, 2, 3, 4, 5, 6, 7, 8, 9, 10, 11, 12, 13, 14, 15);

            acc = __builtin_amdgcn_wmma_f32_16x16x32_f16(
                      false, a, false, b, (short)0, acc, false, false);
        }
        __syncthreads();
    }

    // epilogue: BN + optional residual + ReLU (C/D layout: lanes>=16 hold M 8..15)
    const float sc = bg[nc] * rsqrtf(bv[nc] + 1e-5f);
    const float sh = bb[nc] - bm[nc] * sc;
    const size_t rowbase = ((size_t)n * PH + (y + 1)) * PADW * CCH + (size_t)(xt * 16 + 1) * CCH;
#pragma unroll
    for (int v = 0; v < 8; ++v) {
        const int xloc = v + kb * 8;                 // pixel within tile
        const size_t oidx = rowbase + (size_t)xloc * CCH + nc;
        float val = acc[v] * sc + sh;
        if (res) val += (float)res[oidx];
        val = fmaxf(val, 0.0f);
        out[oidx] = (_Float16)val;
    }
}

// ---------------------------------------------------------------------------
// Kernel 4: write current_feats as f32 NCHW [2,64,48,160] into d_out[0:983040)
// ---------------------------------------------------------------------------
__global__ __launch_bounds__(256)
void writeout_feats(const _Float16* __restrict__ feats, float* __restrict__ out)
{
    const long long idx = (long long)blockIdx.x * 256 + threadIdx.x;
    const long long total = 2LL * CCH * MH * MW;
    if (idx >= total) return;
    const int x = (int)(idx % MW);
    long long t = idx / MW;
    const int y = (int)(t % MH); t /= MH;
    const int c = (int)(t % CCH);
    const int b = (int)(t / CCH);
    out[idx] = (float)feats[((size_t)b * PH + (y + 1)) * PADW * CCH + (size_t)(x + 1) * CCH + c];
}

// ---------------------------------------------------------------------------
// Kernel 5: plane-sweep cost volume [2,80,48,160] from lookup feats (imgs 2,3)
// ---------------------------------------------------------------------------
__global__ __launch_bounds__(256)
void cost_volume(const _Float16* __restrict__ feats,  // padded NHWC f16
                 const float* __restrict__ poses,     // [2,1,4,4]
                 const float* __restrict__ Km,        // [2,4,4]
                 const float* __restrict__ invK,      // [2,4,4]
                 float* __restrict__ outc)            // [2,80,48,160]
{
    const long long idx = (long long)blockIdx.x * 256 + threadIdx.x;
    const long long total = 2LL * NBINS * MH * MW;
    if (idx >= total) return;
    const int x = (int)(idx % MW);
    long long t = idx / MW;
    const int y = (int)(t % MH); t /= MH;
    const int d = (int)(t % NBINS);
    const int b = (int)(t / NBINS);

    const float* ik = invK + b * 16;
    const float* km = Km   + b * 16;
    const float* T  = poses + b * 16;   // NF==1

    const float cx = ik[0] * x + ik[1] * y + ik[2];
    const float cy = ik[4] * x + ik[5] * y + ik[6];
    const float cz = ik[8] * x + ik[9] * y + ik[10];

    float P[3][4];
#pragma unroll
    for (int r = 0; r < 3; ++r)
#pragma unroll
        for (int c = 0; c < 4; ++c)
            P[r][c] = km[r * 4 + 0] * T[c]     + km[r * 4 + 1] * T[4 + c]
                    + km[r * 4 + 2] * T[8 + c] + km[r * 4 + 3] * T[12 + c];

    const float depth = (float)d * (112.0f / 79.0f);   // linspace(0,112,80)
    const float vx = depth * (P[0][0] * cx + P[0][1] * cy + P[0][2] * cz) + P[0][3];
    const float vy = depth * (P[1][0] * cx + P[1][1] * cy + P[1][2] * cz) + P[1][3];
    const float vz = depth * (P[2][0] * cx + P[2][1] * cy + P[2][2] * cz) + P[2][3];
    const float ix = vx / (vz + 1e-7f);
    const float iy = vy / (vz + 1e-7f);

    const float x0 = floorf(ix), y0 = floorf(iy);
    const float wx = ix - x0,    wy = iy - y0;

    const _Float16* fb = feats + (size_t)(2 + b) * PIMG;

    float accum = 0.0f;
#pragma unroll
    for (int tap = 0; tap < 4; ++tap) {
        const float xi = x0 + (float)(tap & 1);
        const float yi = y0 + (float)(tap >> 1);
        const float wgt = ((tap & 1) ? wx : 1.0f - wx) * ((tap >> 1) ? wy : 1.0f - wy);
        const bool valid = (xi >= 0.0f) && (xi <= (float)(MW - 1)) &&
                           (yi >= 0.0f) && (yi <= (float)(MH - 1));
        if (!valid) continue;
        const int xc = (int)xi, yc = (int)yi;
        const v8h* p = (const v8h*)(fb + ((size_t)(yc + 1) * PADW + (xc + 1)) * CCH);
        float s = 0.0f;
#pragma unroll
        for (int j = 0; j < 8; ++j) {
            const v8h hv = p[j];
#pragma unroll
            for (int c = 0; c < 8; ++c) s += (float)hv[c];
        }
        accum += wgt * s * (1.0f / 64.0f);
    }
    outc[idx] = accum;   // idx is already [b,d,y,x] row-major
}

// ---------------------------------------------------------------------------
// Host launcher
// ---------------------------------------------------------------------------
extern "C" void kernel_launch(void* const* d_in, const int* in_sizes, int n_in,
                              void* d_out, int out_size, void* d_ws, size_t ws_size,
                              hipStream_t stream) {
    (void)n_in; (void)out_size; (void)ws_size;
    auto F = [&](int i) { return (const float*)d_in[i]; };

    // ---- decode top-level ordering (insertion vs sorted-key flatten) ----
    const float *cur, *lkp, *poses, *Km, *invK;
    int base;
    const bool top_sorted = (in_sizes[0] != 2 * 3 * IMG_H * IMG_W);
    if (!top_sorted) {
        cur = F(0); lkp = F(1); poses = F(2); Km = F(3); invK = F(4); base = 5;
    } else {
        Km = F(0); cur = F(1); invK = F(2); lkp = F(3); base = 4; poses = F(29);
    }

    // ---- decode params ordering (sorted keys vs insertion order) ----
    const float *c1w, *s_g, *s_b, *s_m, *s_v;
    struct BlkP { const float *w1, *g1, *b1, *m1, *v1, *w2, *g2, *b2, *m2, *v2; } blk[2];
    if (in_sizes[base] == 64) {
        // sorted: bn1(4), block1{bn1(4),bn2(4),conv1,conv2}, block2{...}, conv1
        s_g = F(base + 0); s_b = F(base + 1); s_m = F(base + 2); s_v = F(base + 3);
        blk[0] = { F(base + 12), F(base + 4), F(base + 5), F(base + 6), F(base + 7),
                   F(base + 13), F(base + 8), F(base + 9), F(base + 10), F(base + 11) };
        blk[1] = { F(base + 22), F(base + 14), F(base + 15), F(base + 16), F(base + 17),
                   F(base + 23), F(base + 18), F(base + 19), F(base + 20), F(base + 21) };
        c1w = F(base + 24);
    } else {
        // insertion: conv1, bn1(4), block1{conv1,bn1(4),conv2,bn2(4)}, block2{...}
        c1w = F(base + 0);
        s_g = F(base + 1); s_b = F(base + 2); s_m = F(base + 3); s_v = F(base + 4);
        blk[0] = { F(base + 5),  F(base + 6),  F(base + 7),  F(base + 8),  F(base + 9),
                   F(base + 10), F(base + 11), F(base + 12), F(base + 13), F(base + 14) };
        blk[1] = { F(base + 15), F(base + 16), F(base + 17), F(base + 18), F(base + 19),
                   F(base + 20), F(base + 21), F(base + 22), F(base + 23), F(base + 24) };
    }

    // ---- workspace layout ----
    _Float16* f0h  = (_Float16*)d_ws;                                // [4,96,320,64]
    _Float16* buf0 = (_Float16*)((char*)d_ws + 15728640);            // padded bufs
    _Float16* buf1 = (_Float16*)((char*)d_ws + 15728640 + 1 * PBUF_BYTES);
    _Float16* buf2 = (_Float16*)((char*)d_ws + 15728640 + 2 * PBUF_BYTES);
    _Float16* pwb  = (_Float16*)((char*)d_ws + 15728640 + 3 * PBUF_BYTES);

    // zero halos (and interiors, cheap) of the three padded buffers
    {
        const long long nvec = 3LL * PBUF_BYTES / 16;
        zero_bufs<<<(unsigned)((nvec + 255) / 256), 256, 0, stream>>>((v8h*)buf0, nvec);
    }

    _Float16* pw[4] = { pwb, pwb + PW_ELEMS, pwb + 2 * PW_ELEMS, pwb + 3 * PW_ELEMS };
    const float* rawW[4] = { blk[0].w1, blk[0].w2, blk[1].w1, blk[1].w2 };
    for (int l = 0; l < 4; ++l)
        pack_weights<<<(PW_ELEMS + 255) / 256, 256, 0, stream>>>(rawW[l], pw[l]);

    // layer0 + maxpool
    conv1_bn_relu<<<dim3(F0_W / 4, F0_H, NIMG), 256, 0, stream>>>(
        cur, lkp, c1w, s_g, s_b, s_m, s_v, f0h);
    {
        const long long total = (long long)NIMG * MH * MW * CCH;
        maxpool3x3s2<<<(unsigned)((total + 255) / 256), 256, 0, stream>>>(f0h, buf0);
    }

    // BasicBlocks (padded in/out)
    dim3 cgrid(MW / 16, MH, NIMG);
    conv3x3_wmma_bn<<<cgrid, 128, 0, stream>>>(buf0, pw[0], blk[0].g1, blk[0].b1,
                                               blk[0].m1, blk[0].v1, nullptr, buf1);
    conv3x3_wmma_bn<<<cgrid, 128, 0, stream>>>(buf1, pw[1], blk[0].g2, blk[0].b2,
                                               blk[0].m2, blk[0].v2, buf0, buf2);
    conv3x3_wmma_bn<<<cgrid, 128, 0, stream>>>(buf2, pw[2], blk[1].g1, blk[1].b1,
                                               blk[1].m1, blk[1].v1, nullptr, buf0);
    conv3x3_wmma_bn<<<cgrid, 128, 0, stream>>>(buf0, pw[3], blk[1].g2, blk[1].b2,
                                               blk[1].m2, blk[1].v2, buf2, buf1);
    // buf1 = final feats, padded (images 0,1 = current; 2,3 = lookup)

    writeout_feats<<<(2 * CCH * MH * MW + 255) / 256, 256, 0, stream>>>(
        buf1, (float*)d_out);
    cost_volume<<<(2 * NBINS * MH * MW + 255) / 256, 256, 0, stream>>>(
        buf1, poses, Km, invK, (float*)d_out + 2 * CCH * MH * MW);
}
